// LLAMABlock_13297218749128
// MI455X (gfx1250) — matrix operand order repaired and checked by reading, not compiled, not run on hardware
//
#include <hip/hip_runtime.h>
#include <hip/hip_bf16.h>
#include <math.h>

#if __has_include(<hip/amd_detail/amd_gfx1250_TDM.h>)
#define TDM_6ARG 1
#endif

// ---------------------------------------------------------------------------
// LLaMA block for MI455X (gfx1250): bf16 WMMA everywhere, f32 accumulate.
// GEMM W-tiles are streamed into LDS by the Tensor Data Mover (TDM),
// double-buffered on TENSORcnt. Shapes hardcoded: B=2,T=2048,D=1024,H=16,
// hd=64, FF=4096.
// ---------------------------------------------------------------------------

typedef __bf16 bf16_t;
typedef __attribute__((ext_vector_type(16))) __bf16 v16bf;
typedef __attribute__((ext_vector_type(8)))  __bf16 v8bf;
typedef __attribute__((ext_vector_type(8)))  float  v8f;
typedef __attribute__((ext_vector_type(4)))  unsigned v4u;
typedef __attribute__((ext_vector_type(8)))  int      v8i;
typedef __attribute__((ext_vector_type(4)))  int      v4i;

#define D_MODEL 1024
#define T_SEQ   2048
#define BATCH   2
#define NHEAD   16
#define HD      64
#define FF      4096
#define MROWS   (BATCH * T_SEQ)   // 4096 token rows

__device__ __forceinline__ bf16_t f2bf(float f) {
  unsigned u = __builtin_bit_cast(unsigned, f);
  unsigned r = (u + 0x7FFFu + ((u >> 16) & 1u)) >> 16;   // round-to-nearest-even
  unsigned short h = (unsigned short)r;
  return __builtin_bit_cast(bf16_t, h);
}

__device__ __forceinline__ v8f wmma_bf16(v16bf a, v16bf b, v8f c) {
  // D = A(16x32 bf16) * B(32x16 bf16) + C(16x16 f32)
  return __builtin_amdgcn_wmma_f32_16x16x32_bf16(false, a, false, b,
                                                 (short)0, c, false, false);
}

__device__ __forceinline__ v16bf join16(v8bf lo, v8bf hh) {
  v16bf r;
#pragma unroll
  for (int i = 0; i < 8; ++i) { r[i] = lo[i]; r[8 + i] = hh[i]; }
  return r;
}

// Issue a TDM load of a 2D bf16 tile: rows*cols elements from src (row stride
// srcK elements) into LDS at ldsPtr, with 16B padding appended per 64B row
// (LDS row pitch = 80B) to spread B-fragment reads across banks.
__device__ __forceinline__ void tdm_load_tile_256x32(const bf16_t* src, int srcK, int srcN,
                                                     void* ldsPtr) {
  unsigned lds = (unsigned)(unsigned long long)ldsPtr;  // low 32 bits = LDS offset
  unsigned long long ga = (unsigned long long)src;
  v4u g0;
  g0[0] = 1u;                                        // count=1: valid descriptor
  g0[1] = lds;                                       // lds_addr
  g0[2] = (unsigned)(ga & 0xFFFFFFFFu);              // global_addr[31:0]
  g0[3] = (unsigned)((ga >> 32) & 0x1FFFFFFu) | (2u << 30);  // addr[56:32] | type=2
  v8i g1;
  g1[0] = (int)((1u << 16) |                         // data_size = 2 bytes
                (1u << 20) |                         // pad_enable
                (3u << 22) |                         // pad_interval: 16 DWORDs (64B row)
                (3u << 25));                         // pad_amount: 4 DWORDs (16B)
  g1[1] = (int)(((unsigned)srcK & 0xFFFFu) << 16);   // tensor_dim0[15:0]
  g1[2] = (int)((((unsigned)srcK >> 16) & 0xFFFFu) |
                (((unsigned)srcN & 0xFFFFu) << 16)); // dim0 hi | tensor_dim1 lo
  g1[3] = (int)(((unsigned)srcN >> 16) & 0xFFFFu) | (int)(32u << 16); // dim1 hi | tile_dim0=32
  g1[4] = (int)256;                                  // tile_dim1=256, tile_dim2=0
  g1[5] = (int)srcK;                                 // tensor_dim0_stride[31:0] (elements)
  g1[6] = 0;                                         // stride hi | dim1_stride (unused, 2D)
  g1[7] = 0;
  v4i z = {0, 0, 0, 0};
#ifdef TDM_6ARG
  v8i z8 = {0, 0, 0, 0, 0, 0, 0, 0};
  __builtin_amdgcn_tensor_load_to_lds(g0, g1, z, z, z8, 0);
#else
  __builtin_amdgcn_tensor_load_to_lds(g0, g1, z, z, 0);
#endif
}

// ---------------------------------------------------------------------------
// fp32 -> bf16 tensor conversion (weights), 4 elems/thread
// ---------------------------------------------------------------------------
__global__ void cvt_f32_bf16(const float* __restrict__ in, bf16_t* __restrict__ out, int n) {
  int i = (blockIdx.x * blockDim.x + threadIdx.x) * 4;
  if (i + 3 < n) {
    float4 v = *(const float4*)(in + i);
    out[i + 0] = f2bf(v.x);
    out[i + 1] = f2bf(v.y);
    out[i + 2] = f2bf(v.z);
    out[i + 3] = f2bf(v.w);
  }
}

// ---------------------------------------------------------------------------
// RMSNorm (+ interleaved RoPE) over one D=1024 row per block.
// ---------------------------------------------------------------------------
__global__ void rmsnorm_rope_kernel(const float* __restrict__ x,
                                    const float* __restrict__ g,
                                    bf16_t* __restrict__ qk,
                                    bf16_t* __restrict__ vv) {
  const int row = blockIdx.x;           // b*T + t
  const int t   = row & (T_SEQ - 1);
  const int tid = threadIdx.x;
  const float* xr = x + (size_t)row * D_MODEL;
  float4 v = *(const float4*)(xr + tid * 4);
  float ss = v.x * v.x + v.y * v.y + v.z * v.z + v.w * v.w;
#pragma unroll
  for (int m = 16; m >= 1; m >>= 1) ss += __shfl_xor(ss, m, 32);
  __shared__ float red[8];
  if ((tid & 31) == 0) red[tid >> 5] = ss;
  __syncthreads();
  float tot = 0.f;
#pragma unroll
  for (int i = 0; i < 8; ++i) tot += red[i];
  const float inv = rsqrtf(tot * (1.0f / D_MODEL) + 1e-6f);

  const int c0 = tid * 4;
  float xn[4] = { v.x * inv * g[c0 + 0], v.y * inv * g[c0 + 1],
                  v.z * inv * g[c0 + 2], v.w * inv * g[c0 + 3] };
#pragma unroll
  for (int i = 0; i < 4; ++i) vv[(size_t)row * D_MODEL + c0 + i] = f2bf(xn[i]);
#pragma unroll
  for (int p = 0; p < 2; ++p) {         // two interleaved pairs per thread
    const int i0 = c0 + 2 * p;
    const int pairi = (i0 & (HD - 1)) >> 1;
    const float theta = __powf(10000.0f, -(float)(2 * pairi) * (1.0f / HD));
    float sn, cs;
    __sincosf((float)t * theta, &sn, &cs);
    const float x0 = xn[2 * p], x1 = xn[2 * p + 1];
    qk[(size_t)row * D_MODEL + i0 + 0] = f2bf(x0 * cs - x1 * sn);
    qk[(size_t)row * D_MODEL + i0 + 1] = f2bf(x1 * cs + x0 * sn);
  }
}

__global__ void rmsnorm_kernel(const float* __restrict__ x,
                               const float* __restrict__ g,
                               bf16_t* __restrict__ out) {
  const int row = blockIdx.x;
  const int tid = threadIdx.x;
  const float* xr = x + (size_t)row * D_MODEL;
  float4 v = *(const float4*)(xr + tid * 4);
  float ss = v.x * v.x + v.y * v.y + v.z * v.z + v.w * v.w;
#pragma unroll
  for (int m = 16; m >= 1; m >>= 1) ss += __shfl_xor(ss, m, 32);
  __shared__ float red[8];
  if ((tid & 31) == 0) red[tid >> 5] = ss;
  __syncthreads();
  float tot = 0.f;
#pragma unroll
  for (int i = 0; i < 8; ++i) tot += red[i];
  const float inv = rsqrtf(tot * (1.0f / D_MODEL) + 1e-6f);
  const int c0 = tid * 4;
  out[(size_t)row * D_MODEL + c0 + 0] = f2bf(v.x * inv * g[c0 + 0]);
  out[(size_t)row * D_MODEL + c0 + 1] = f2bf(v.y * inv * g[c0 + 1]);
  out[(size_t)row * D_MODEL + c0 + 2] = f2bf(v.z * inv * g[c0 + 2]);
  out[(size_t)row * D_MODEL + c0 + 3] = f2bf(v.w * inv * g[c0 + 3]);
}

// ---------------------------------------------------------------------------
// Generic WMMA GEMM:  C[M,N] = epilogue(A[M,K] @ W[N,K]^T + bias[N])
// 8 waves in a 2x4 grid; each wave owns a 32(M) x 64(N) tile: 2 A-fragments x
// 4 B-fragments -> 8 v_wmma per 32-wide K chunk. The 256x32 W tile for the
// whole block is DMA'd into LDS by the TDM (wave 0 issues tensor_load_to_lds,
// double-buffered on TENSORcnt; two barriers per chunk), so each lane's
// B-fragment is two 16B ds_loads from a bank-padded 80B-pitch row, and W
// global traffic per block drops ~8x vs per-lane loads.
// Block tile: 64(M) x 256(N).  grid = (N/256, M/64).
// MODE: 0=f32 store, 1=SiLU->bf16, 2=QKV scatter [B,H,T,hd] bf16, 3=resid add f32
// ---------------------------------------------------------------------------
template <int N, int K, int MODE>
__global__ __launch_bounds__(256, 2)
void gemm_wmma(const bf16_t* __restrict__ A,
               const bf16_t* __restrict__ W,
               const float*  __restrict__ bias,
               void*         __restrict__ out,
               const float*  __restrict__ resid) {
  __shared__ alignas(16) bf16_t Wt[2][256][40];   // 2 x 20KB, 80B row pitch (TDM-padded)
  const int tid  = threadIdx.x, lane = tid & 31, wid = tid >> 5;
  const int wm = wid >> 2, wn = wid & 3;
  const int m0 = blockIdx.y * 64 + wm * 32;
  const int n0blk = blockIdx.x * 256;
  const int arow = lane & 15;
  const int hi   = (lane >= 16);

  v8f acc[2][4] = {};
  const bf16_t* aptr0 = A + (size_t)(m0 + arow) * K + (hi ? 8 : 0);
  const bf16_t* aptr1 = aptr0 + (size_t)16 * K;
  const bf16_t* wtile = W + (size_t)n0blk * K;

  if (wid == 0) tdm_load_tile_256x32(wtile, K, N, &Wt[0][0][0]);

  const int nchunks = K / 32;
  for (int i = 0; i < nchunks; ++i) {
    const int k0 = i * 32;
    if (wid == 0) __builtin_amdgcn_s_wait_tensorcnt(0);   // current buffer landed
    __syncthreads();                                      // publish to all waves
    if (wid == 0 && i + 1 < nchunks)                      // overlap next DMA w/ compute
      tdm_load_tile_256x32(wtile + k0 + 32, K, N, &Wt[(i + 1) & 1][0][0]);

    v16bf af0 = join16(*(const v8bf*)(aptr0 + k0), *(const v8bf*)(aptr0 + k0 + 16));
    v16bf af1 = join16(*(const v8bf*)(aptr1 + k0), *(const v8bf*)(aptr1 + k0 + 16));
    __builtin_prefetch(aptr0 + k0 + 64, 0, 1);   // global_prefetch next A chunks
    __builtin_prefetch(aptr1 + k0 + 64, 0, 1);

    const bf16_t (*wb)[40] = &Wt[i & 1][wn * 64];
#pragma unroll
    for (int ns = 0; ns < 4; ++ns) {
      const bf16_t* wp = &wb[ns * 16 + arow][hi ? 8 : 0];           // 16B aligned
      v16bf bfrag = join16(*(const v8bf*)(wp), *(const v8bf*)(wp + 16));
      acc[0][ns] = wmma_bf16(af0, bfrag, acc[0][ns]);
      acc[1][ns] = wmma_bf16(af1, bfrag, acc[1][ns]);
    }
    __syncthreads();   // all reads of this buffer done before it is re-filled
  }

  const int rbase = hi ? 8 : 0;
#pragma unroll
  for (int mi = 0; mi < 2; ++mi) {
#pragma unroll
    for (int ns = 0; ns < 4; ++ns) {
      const int col = n0blk + (wid & 3) * 64 + ns * 16 + arow;
      const float bcol = bias[col];
#pragma unroll
      for (int r = 0; r < 8; ++r) {
        const int row = m0 + mi * 16 + rbase + r;
        float vv = acc[mi][ns][r] + bcol;
        if (MODE == 0) {
          ((float*)out)[(size_t)row * N + col] = vv;
        } else if (MODE == 1) {
          float s = vv / (1.0f + __expf(-vv));     // SiLU
          ((bf16_t*)out)[(size_t)row * N + col] = f2bf(s);
        } else if (MODE == 2) {                    // [B,T]x[D] -> [B,H,T,hd]
          const int bb = row >> 11, tt = row & (T_SEQ - 1);
          const int hh = col >> 6, dd = col & (HD - 1);
          ((bf16_t*)out)[(((size_t)bb * NHEAD + hh) * T_SEQ + tt) * HD + dd] = f2bf(vv);
        } else {                                   // residual add, final output
          ((float*)out)[(size_t)row * N + col] = resid[(size_t)row * N + col] + vv;
        }
      }
    }
  }
}

// ---------------------------------------------------------------------------
// Flash attention: grid (T/128, B*H), 256 threads = 8 waves x 16 query rows.
// Q/K/V in [B*H, T, hd] bf16.  Online softmax with scale 1/8, causal mask.
// (unchanged from round 2: spill-free with __launch_bounds__(256,1))
// ---------------------------------------------------------------------------
__global__ __launch_bounds__(256, 1)
void attn_kernel(const bf16_t* __restrict__ Q,
                 const bf16_t* __restrict__ Kx,
                 const bf16_t* __restrict__ V,
                 bf16_t* __restrict__ ctx) {
  const int bh = blockIdx.y;
  const int q0 = blockIdx.x * 128;
  const int tid = threadIdx.x, lane = tid & 31, wave = tid >> 5;
  const int b = bh >> 4, h = bh & 15;
  const int arow = lane & 15;
  const int hi   = (lane >= 16);

  __shared__ alignas(16) bf16_t Vt[HD][40];   // V chunk transposed; 80B row pitch
  __shared__ alignas(16) bf16_t P[8][16][32]; // per-wave softmax tile relayout buffer

  const size_t base = (size_t)bh * T_SEQ * HD;
  const int qrow0 = q0 + wave * 16;

  v16bf qa0, qa1;
  {
    const bf16_t* qp = Q + base + (size_t)(qrow0 + arow) * HD + (hi ? 16 : 0);
    qa0 = join16(*(const v8bf*)(qp),      *(const v8bf*)(qp + 8));
    qa1 = join16(*(const v8bf*)(qp + 32), *(const v8bf*)(qp + 40));
  }

  v8f o0 = {}, o1 = {}, o2 = {}, o3 = {};
  float mrow[8], lrow[8];
#pragma unroll
  for (int r = 0; r < 8; ++r) { mrow[r] = -1e30f; lrow[r] = 0.f; }

  const int kmax = q0 + 128;             // causal: keys <= last row of block
  for (int kk = 0; kk < kmax; kk += 32) {
    __syncthreads();                     // Vt reads of prev iter done before restage
    {                                    // stage V chunk transposed: Vt[d][key]
      const int key = tid >> 3, d0 = (tid & 7) * 8;
      v8bf vv = *(const v8bf*)(V + base + (size_t)(kk + key) * HD + d0);
#pragma unroll
      for (int j = 0; j < 8; ++j) Vt[d0 + j][key] = vv[j];
    }
    __syncthreads();

    float st[2][8];
#pragma unroll
    for (int jt = 0; jt < 2; ++jt) {
      const int j = jt * 16;
      const bf16_t* kp = Kx + base + (size_t)(kk + j + arow) * HD + (hi ? 16 : 0);
      v16bf kb0 = *(const v16bf*)(kp);        // d 0..15 / 16..31
      v16bf kb1 = *(const v16bf*)(kp + 32);   // d 32..47 / 48..63
      v8f s = {};
      s = wmma_bf16(qa0, kb0, s);
      s = wmma_bf16(qa1, kb1, s);
      const int keyabs = kk + j + arow;
#pragma unroll
      for (int r = 0; r < 8; ++r) {
        float sv = s[r] * 0.125f;             // 1/sqrt(64)
        if (keyabs > qrow0 + (hi ? 8 : 0) + r) sv = -1e30f;   // causal
        st[jt][r] = sv;
      }
    }

#pragma unroll
    for (int r = 0; r < 8; ++r) {
      float cm = fmaxf(st[0][r], st[1][r]);
#pragma unroll
      for (int m = 8; m >= 1; m >>= 1) cm = fmaxf(cm, __shfl_xor(cm, m, 16));
      const float nm = fmaxf(mrow[r], cm);
      const float sc = __expf(mrow[r] - nm);
      mrow[r] = nm;
      const float p0 = __expf(st[0][r] - nm);
      const float p1 = __expf(st[1][r] - nm);
      float rs = p0 + p1;
#pragma unroll
      for (int m = 8; m >= 1; m >>= 1) rs += __shfl_xor(rs, m, 16);
      lrow[r] = lrow[r] * sc + rs;
      o0[r] *= sc; o1[r] *= sc; o2[r] *= sc; o3[r] *= sc;
      const int rr = (hi ? 8 : 0) + r;        // C-tile row -> P row
      P[wave][rr][arow]      = f2bf(p0);
      P[wave][rr][16 + arow] = f2bf(p1);
    }

    const bf16_t* pp = &P[wave][arow][hi ? 8 : 0];   // 16B-aligned
    v16bf pa = join16(*(const v8bf*)(pp), *(const v8bf*)(pp + 16));

#pragma unroll
    for (int n = 0; n < 4; ++n) {
      const bf16_t* vtp = &Vt[n * 16 + arow][hi ? 8 : 0];    // 16B-aligned
      v16bf vb = join16(*(const v8bf*)(vtp), *(const v8bf*)(vtp + 16));
      v8f& oo = (n == 0) ? o0 : (n == 1) ? o1 : (n == 2) ? o2 : o3;
      oo = wmma_bf16(pa, vb, oo);
    }
  }

#pragma unroll
  for (int n = 0; n < 4; ++n) {
    const v8f& oo = (n == 0) ? o0 : (n == 1) ? o1 : (n == 2) ? o2 : o3;
    const int d = n * 16 + arow;
#pragma unroll
    for (int r = 0; r < 8; ++r) {
      const int row = qrow0 + (hi ? 8 : 0) + r;
      const float val = oo[r] / lrow[r];
      ctx[(((size_t)b * T_SEQ + row) * NHEAD + h) * HD + d] = f2bf(val);
    }
  }
}

// ---------------------------------------------------------------------------
// Host-side orchestration
// ---------------------------------------------------------------------------
extern "C" void kernel_launch(void* const* d_in, const int* in_sizes, int n_in,
                              void* d_out, int out_size, void* d_ws, size_t ws_size,
                              hipStream_t stream) {
  (void)in_sizes; (void)n_in; (void)out_size; (void)ws_size;
  const float* x      = (const float*)d_in[0];
  const float* gscale = (const float*)d_in[1];
  const float* in_w   = (const float*)d_in[2];
  const float* in_b   = (const float*)d_in[3];
  const float* out_w  = (const float*)d_in[4];
  const float* out_b  = (const float*)d_in[5];
  const float* w1     = (const float*)d_in[6];
  const float* b1     = (const float*)d_in[7];
  const float* w2     = (const float*)d_in[8];
  const float* b2     = (const float*)d_in[9];
  float* out = (float*)d_out;
  char*  ws  = (char*)d_ws;

  // Workspace layout (bytes). Total ~92 MB with aliasing.
  bf16_t* wqkv = (bf16_t*)(ws + 0);           // 3*D*D bf16   = 6291456
  bf16_t* wo   = (bf16_t*)(ws + 6291456);     // D*D bf16     = 2097152
  bf16_t* w1b  = (bf16_t*)(ws + 8388608);     // FF*D bf16    = 8388608
  bf16_t* w2b  = (bf16_t*)(ws + 16777216);    // D*FF bf16    = 8388608
  bf16_t* qkin = (bf16_t*)(ws + 25165824);    // M*D bf16     = 8388608
  bf16_t* vin  = (bf16_t*)(ws + 33554432);    // M*D bf16     = 8388608
  bf16_t* q    = (bf16_t*)(ws + 41943040);    // M*D bf16
  bf16_t* k    = (bf16_t*)(ws + 50331648);    // M*D bf16
  bf16_t* v    = (bf16_t*)(ws + 58720256);    // M*D bf16
  bf16_t* ctx  = (bf16_t*)(ws + 67108864);    // M*D bf16
  float*  y    = (float*)(ws + 75497472);     // M*D f32      = 16777216 (ends 92274688)
  bf16_t* h1   = (bf16_t*)(ws + 41943040);    // M*FF bf16: reuses q/k/v/ctx (32MB)
  bf16_t* yn   = qkin;                        // reuses qk_in

  dim3 b256(256);

  // 1) weights -> bf16
  cvt_f32_bf16<<<(3 * D_MODEL * D_MODEL) / 1024, b256, 0, stream>>>(in_w, wqkv, 3 * D_MODEL * D_MODEL);
  cvt_f32_bf16<<<(D_MODEL * D_MODEL) / 1024, b256, 0, stream>>>(out_w, wo, D_MODEL * D_MODEL);
  cvt_f32_bf16<<<(FF * D_MODEL) / 1024, b256, 0, stream>>>(w1, w1b, FF * D_MODEL);
  cvt_f32_bf16<<<(D_MODEL * FF) / 1024, b256, 0, stream>>>(w2, w2b, D_MODEL * FF);

  // 2) RMSNorm + RoPE (q_in == k_in in the reference)
  rmsnorm_rope_kernel<<<MROWS, b256, 0, stream>>>(x, gscale, qkin, vin);

  // 3) QKV projections, scattering to [B,H,T,hd]
  gemm_wmma<D_MODEL, D_MODEL, 2><<<dim3(4, 64), b256, 0, stream>>>(qkin, wqkv, in_b, q, nullptr);
  gemm_wmma<D_MODEL, D_MODEL, 2><<<dim3(4, 64), b256, 0, stream>>>(qkin, wqkv + D_MODEL * D_MODEL, in_b + D_MODEL, k, nullptr);
  gemm_wmma<D_MODEL, D_MODEL, 2><<<dim3(4, 64), b256, 0, stream>>>(vin, wqkv + 2 * D_MODEL * D_MODEL, in_b + 2 * D_MODEL, v, nullptr);

  // 4) causal flash attention -> ctx [B,T,D]
  attn_kernel<<<dim3(T_SEQ / 128, BATCH * NHEAD), b256, 0, stream>>>(q, k, v, ctx);

  // 5) out-proj -> y (f32, kept for residual)
  gemm_wmma<D_MODEL, D_MODEL, 0><<<dim3(4, 64), b256, 0, stream>>>(ctx, wo, out_b, y, nullptr);

  // 6) RMSNorm -> yn bf16
  rmsnorm_kernel<<<MROWS, b256, 0, stream>>>(y, gscale, yn);

  // 7) FFN1 with SiLU -> h1 bf16
  gemm_wmma<FF, D_MODEL, 1><<<dim3(16, 64), b256, 0, stream>>>(yn, w1b, b1, h1, nullptr);

  // 8) FFN2 + residual y -> d_out (f32)
  gemm_wmma<D_MODEL, FF, 3><<<dim3(4, 64), b256, 0, stream>>>(h1, w2b, b2, out, y);
}
